// Net_57913339019585
// MI455X (gfx1250) — matrix operand order repaired and checked by reading, not compiled, not run on hardware
//
#include <hip/hip_runtime.h>
#include <math.h>

// ---------------------------------------------------------------------------
// Fused persistent SNN forward for gfx1250 (MI455X).
//   T=500, B=1024, IN=4, H=128, OUT=2.
//   One block = 16 batch rows, 8 waves (wave32). Entire time loop in-kernel.
//   Hidden GEMM via v_wmma_f32_16x16x32_f16 with f16 hi/lo weight splitting
//   (B-fragments live in registers for all 500 steps). Layer-3 reduction via
//   DPP butterfly (no LDS round trips). 2 LDS-only barriers per step
//   (s_wait_dscnt + s_barrier, no storecnt/loadcnt drain on the barrier).
// ---------------------------------------------------------------------------

typedef __attribute__((ext_vector_type(16))) _Float16 v16h;
typedef __attribute__((ext_vector_type(8)))  _Float16 v8h;
typedef __attribute__((ext_vector_type(8)))  float    v8f;
typedef __attribute__((ext_vector_type(4)))  float    v4f;
typedef __attribute__((ext_vector_type(2)))  float    v2f;

#define WMMA_F16(A, B, C) \
  __builtin_amdgcn_wmma_f32_16x16x32_f16(false, (A), false, (B), (short)0, (C), false, false)

constexpr int T_STEPS  = 500;
constexpr int BATCH    = 1024;
constexpr int IN_F     = 4;
constexpr int H_F      = 128;
constexpr int OUT_F    = 2;
constexpr int MB       = 16;    // batch rows per block
constexpr int NTHREADS = 256;   // 8 waves of 32

__device__ __forceinline__ float clip01(float v) {
  return fminf(fmaxf(v, 0.0f), 1.0f);
}

// Workgroup barrier that drains ONLY the LDS counter (DScnt). Global loads
// (x prefetch) and global stores (softmax output) are intentionally NOT
// drained here -- they have no cross-wave ordering requirement, and draining
// them (as __syncthreads does) would put global memory latency on the
// 500-step serial critical path.
__device__ __forceinline__ void block_sync_lds() {
  asm volatile(
      "s_wait_dscnt 0x0\n\t"
      "s_barrier_signal -1\n\t"
      "s_barrier_wait -1"
      ::: "memory");
}

// v += lane(v ^ pattern)  via DPP (VALU only, no LDS)
template <int CTRL>
__device__ __forceinline__ float dppAdd(float v) {
  const int x = __builtin_amdgcn_update_dpp(0, __float_as_int(v), CTRL,
                                            0xF, 0xF, true);
  return v + __int_as_float(x);
}

// full butterfly sum across each 16-lane row-group (result in every lane)
__device__ __forceinline__ float rowSum16(float v) {
  v = dppAdd<0xB1>(v);    // quad_perm [1,0,3,2]  -> xor 1
  v = dppAdd<0x4E>(v);    // quad_perm [2,3,0,1]  -> xor 2
  v = dppAdd<0x164>(v);   // ROW_XMASK:4         -> xor 4
  v = dppAdd<0x168>(v);   // ROW_XMASK:8         -> xor 8
  return v;
}

__launch_bounds__(NTHREADS)
__global__ void snn_fused(const float* __restrict__ xg,        // [T,B,IN]
                          const float* __restrict__ W_in,      // [H,IN]
                          const float* __restrict__ beta_in,   // [H]
                          const float* __restrict__ thr_in,    // [H]
                          const float* __restrict__ W_hid,     // [H,H]
                          const float* __restrict__ beta_hid,  // [H]
                          const float* __restrict__ thr_hid,   // [H]
                          const float* __restrict__ W_out,     // [OUT,H]
                          const float* __restrict__ beta_out,  // [1]
                          float* __restrict__ out)             // [T,B,OUT]
{
  __shared__ _Float16 spk_lds[MB][H_F];       // 4 KB  spike tile (f16, exact {0,1})
  __shared__ float    xs[MB * IN_F];          // 256 B input tile
  __shared__ float    partial[8][MB][OUT_F];  // 1 KB  layer-3 per-wave partials

  const int tid  = threadIdx.x;
  const int lane = tid & 31;
  const int wv   = tid >> 5;            // wave id 0..7 -> owns N-tile [16*wv, 16*wv+16)
  const int b0   = blockIdx.x * MB;     // batch base of this block

  // ---------------- layer-1 per-thread state --------------------------------
  // thread t: batch row bb = t&15, hidden rows hb..hb+7 (hb = (t>>4)*8)
  const int bb = tid & 15;
  const int hb = (tid >> 4) * 8;
  v4f   wiv[8];
  float b1v[8], t1v[8], mem1[8];
#pragma unroll
  for (int j = 0; j < 8; ++j) {
    wiv[j]  = *(const v4f*)(W_in + (hb + j) * IN_F);
    b1v[j]  = clip01(beta_in[hb + j]);
    t1v[j]  = thr_in[hb + j];
    mem1[j] = 0.0f;
  }

  // ---------------- layer-2 state in WMMA C/D fragment layout ----------------
  // VGPR r: lanes 0-15 -> M=r, lanes 16-31 -> M=r+8 ; N = 16*wv + (lane&15)
  const int   n2    = wv * 16 + (lane & 15);
  const float beta2 = clip01(beta_hid[n2]);
  const float thr2  = thr_hid[n2];
  const float wo0   = W_out[n2];            // W_out[0][n2]
  const float wo1   = W_out[H_F + n2];      // W_out[1][n2]
  v8f mem2 = {};

  // ---------------- layer-3 state (threads 0..15, b = tid) ------------------
  float m3a = 0.0f, m3b = 0.0f;
  const float beta3 = clip01(beta_out[0]);

  // ---------------- preload W_hidden B-fragments (time-invariant) ------------
  // B is 32x16 (KxN): lane holds column N = lane&15; elements follow the A K order (table order).
  // B[k][n] = W_hid[n0+n][k] ; split into f16 hi + f16 lo for ~fp32 accuracy.
  v16h bh[4], bl[4];
#pragma unroll
  for (int kk = 0; kk < 4; ++kk) {
    const int k = kk * 32 + (lane >> 4) * 8;
#pragma unroll
    for (int n = 0; n < 16; ++n) {
      const float w  = W_hid[(wv * 16 + (lane & 15)) * H_F + k + (n < 8 ? n : n + 8)];
      const _Float16 hi = (_Float16)w;
      bh[kk][n] = hi;
      bl[kk][n] = (_Float16)(w - (float)hi);
    }
  }

  // A-fragment addressing (16-bit A 16x32 layout):
  // lane<16: elems 0..7 = K base+0..7,  elems 8..15 = K base+16..23
  // lane>=16: +8 on both K ranges.
  const int arow = lane & 15;
  const int koff = (lane >> 4) * 8;

  // software-pipelined x load
  float xr = 0.0f;
  if (tid < MB * IN_F) xr = xg[(size_t)b0 * IN_F + tid];

  for (int t = 0; t < T_STEPS; ++t) {
    if (tid < MB * IN_F) xs[tid] = xr;
    block_sync_lds();  // S1: xs(t) visible; partial(t-1) visible

    // ---- layer-3 finalize for step t-1 (deferred one iteration) ------------
    if (t > 0 && tid < MB) {
      float s0 = 0.0f, s1 = 0.0f;
#pragma unroll
      for (int w = 0; w < 8; ++w) {
        const v2f p = *(const v2f*)&partial[w][tid][0];
        s0 += p.x;
        s1 += p.y;
      }
      m3a = beta3 * m3a + s0;
      m3b = beta3 * m3b + s1;
      const float mx  = fmaxf(m3a, m3b);
      const float e0  = __expf(m3a - mx);
      const float e1  = __expf(m3b - mx);
      const float inv = 1.0f / (e0 + e1);
      const size_t o  = (size_t)(t - 1) * (BATCH * OUT_F) + (size_t)(b0 + tid) * OUT_F;
      out[o]     = e0 * inv;    // fire-and-forget; never drained in-loop
      out[o + 1] = e1 * inv;
    }

    // prefetch next timestep's input (a full step of latency hiding)
    if (tid < MB * IN_F && (t + 1) < T_STEPS)
      xr = xg[(size_t)(t + 1) * (BATCH * IN_F) + (size_t)b0 * IN_F + tid];

    // ---- layer 1: LIF with reset-by-subtraction, emit spikes to LDS --------
    {
      const v4f xv = *(const v4f*)&xs[bb * IN_F];   // one b128 LDS read
      v8h sp;
#pragma unroll
      for (int j = 0; j < 8; ++j) {
        const float cur = wiv[j].x * xv.x + wiv[j].y * xv.y +
                          wiv[j].z * xv.z + wiv[j].w * xv.w;
        const float old = mem1[j];
        const float m   = b1v[j] * old + cur - ((old > t1v[j]) ? t1v[j] : 0.0f);
        mem1[j] = m;
        sp[j]   = (m > t1v[j]) ? (_Float16)1.0f : (_Float16)0.0f;
      }
      *(v8h*)&spk_lds[bb][hb] = sp;                 // one b128 LDS write
    }
    block_sync_lds();  // S2: spikes visible; partial(t-1) reads complete

    // ---- layer 2 GEMM: load all A-fragments first, then dual WMMA chains ---
    v16h a[4];
#pragma unroll
    for (int kk = 0; kk < 4; ++kk) {
      const v8h p0 = *(const v8h*)&spk_lds[arow][kk * 32 + koff];
      const v8h p1 = *(const v8h*)&spk_lds[arow][kk * 32 + 16 + koff];
      a[kk] = __builtin_shufflevector(p0, p1,
          0, 1, 2, 3, 4, 5, 6, 7, 8, 9, 10, 11, 12, 13, 14, 15);
    }
    v8f acc_h = {};
    v8f acc_l = {};
#pragma unroll
    for (int kk = 0; kk < 4; ++kk) {
      acc_h = WMMA_F16(a[kk], bh[kk], acc_h);
      acc_l = WMMA_F16(a[kk], bl[kk], acc_l);
    }

    // ---- layer 2 LIF update + layer-3 partial reduction (DPP butterfly) ----
#pragma unroll
    for (int r = 0; r < 8; ++r) {
      const float cur = acc_h[r] + acc_l[r];
      const float old = mem2[r];
      const float m   = beta2 * old + cur - ((old > thr2) ? thr2 : 0.0f);
      mem2[r] = m;
      const float s = (m > thr2) ? 1.0f : 0.0f;
      float v0 = rowSum16(s * wo0);
      float v1 = rowSum16(s * wo1);
      if ((lane & 15) == 0) {
        const int mrow = r + ((lane & 16) ? 8 : 0);
        v2f p;
        p.x = v0;
        p.y = v1;
        *(v2f*)&partial[wv][mrow][0] = p;
      }
    }
    // no barrier here: partial(t) is consumed after S1 of iteration t+1
  }

  // ---- epilogue: finalize last step ----------------------------------------
  block_sync_lds();
  if (tid < MB) {
    float s0 = 0.0f, s1 = 0.0f;
#pragma unroll
    for (int w = 0; w < 8; ++w) {
      const v2f p = *(const v2f*)&partial[w][tid][0];
      s0 += p.x;
      s1 += p.y;
    }
    m3a = beta3 * m3a + s0;
    m3b = beta3 * m3b + s1;
    const float mx  = fmaxf(m3a, m3b);
    const float e0  = __expf(m3a - mx);
    const float e1  = __expf(m3b - mx);
    const float inv = 1.0f / (e0 + e1);
    const size_t o  = (size_t)(T_STEPS - 1) * (BATCH * OUT_F) + (size_t)(b0 + tid) * OUT_F;
    out[o]     = e0 * inv;
    out[o + 1] = e1 * inv;
  }
}

extern "C" void kernel_launch(void* const* d_in, const int* in_sizes, int n_in,
                              void* d_out, int out_size, void* d_ws, size_t ws_size,
                              hipStream_t stream) {
  (void)in_sizes; (void)n_in; (void)out_size; (void)d_ws; (void)ws_size;
  const float* x        = (const float*)d_in[0];
  const float* W_in     = (const float*)d_in[1];
  const float* beta_in  = (const float*)d_in[2];
  const float* thr_in   = (const float*)d_in[3];
  const float* W_hid    = (const float*)d_in[4];
  const float* beta_hid = (const float*)d_in[5];
  const float* thr_hid  = (const float*)d_in[6];
  const float* W_out    = (const float*)d_in[7];
  const float* beta_out = (const float*)d_in[8];
  float* out = (float*)d_out;

  snn_fused<<<BATCH / MB, NTHREADS, 0, stream>>>(
      x, W_in, beta_in, thr_in, W_hid, beta_hid, thr_hid, W_out, beta_out, out);
}
